// MultiHeadAttention_730144440852
// MI455X (gfx1250) — matrix operand order repaired and checked by reading, not compiled, not run on hardware
//
#include <hip/hip_runtime.h>
#include <hip/hip_bf16.h>
#include <math.h>

#define BATCH 4
#define CDIM  256
#define NPOS  2048
#define LDIM  (3 * NPOS)   // 6144
#define HEADS 8
#define KDH   96           // Ch*3

typedef __attribute__((ext_vector_type(16))) __bf16 v16bf;
typedef __attribute__((ext_vector_type(8)))  __bf16 v8bf;
typedef __attribute__((ext_vector_type(2)))  __bf16 bf16x2;
typedef __attribute__((ext_vector_type(8)))  float  v8f;

#define WMMA_BF16(A, B, C) \
  __builtin_amdgcn_wmma_f32_16x16x32_bf16(false, (A), false, (B), (short)0, (C), false, false)

__device__ __forceinline__ __bf16 f2bf(float f) {
  unsigned u = __builtin_bit_cast(unsigned, f);
  unsigned r = (u + 0x7fffu + ((u >> 16) & 1u)) >> 16;
  return __builtin_bit_cast(__bf16, (unsigned short)r);
}

// 8 contiguous bf16 (16 bytes, must be 16B aligned) -> one global_load_b128.
__device__ __forceinline__ v8bf ld8(const __bf16* p) {
  return __builtin_bit_cast(v8bf, *(const uint4*)p);
}

// ---------------------------------------------------------------------------
// Prep: Wt[e][c] = bf16(W[c][e]) (transposed weight), U[e][d] = EPS*b/||b||.
// One 256-thread block per weight set.
// ---------------------------------------------------------------------------
__global__ void vn_prep(const float* __restrict__ W, const float* __restrict__ Bv,
                        __bf16* __restrict__ Wt, float* __restrict__ U) {
  const int e = threadIdx.x;  // 0..255
  #pragma unroll 4
  for (int c = 0; c < CDIM; ++c) {
    Wt[e * CDIM + c] = f2bf(W[c * CDIM + e]);
  }
  const float b0 = Bv[e * 3 + 0], b1 = Bv[e * 3 + 1], b2 = Bv[e * 3 + 2];
  const float s = 1e-6f * rsqrtf(fmaxf(b0 * b0 + b1 * b1 + b2 * b2, 1e-38f));
  U[e * 3 + 0] = b0 * s;
  U[e * 3 + 1] = b1 * s;
  U[e * 3 + 2] = b2 * s;
}

// ---------------------------------------------------------------------------
// Projection GEMM: y[e,l] = sum_c Wt[e][c] * X[c][l] + U[e][l/NPOS].
// One wave computes a 16(e) x 64(l) tile: the A fragment (weights) is loaded
// once per K-step (2 x b128) and reused for 4 WMMAs. K = 256 in 8 steps.
// MODE 0: bf16 out, row layout [b,h,pos,96]         (for q, k)
// MODE 1: bf16 out, natural   [b,e,d,n]             (for z)
// MODE 2: f32  out, natural   [b,e,d,n]             (final output)
// INBF16: X is bf16 (attention output) vs f32 (original inputs).
// ---------------------------------------------------------------------------
template <int MODE, int INBF16>
__global__ void __launch_bounds__(256) vn_proj(
    const void* __restrict__ Xv, const __bf16* __restrict__ Wt,
    const float* __restrict__ U, void* __restrict__ Yv) {
  const int lane = threadIdx.x & 31;
  const int wave = threadIdx.x >> 5;
  const int ltiles = LDIM / 64;   // 96
  const int etiles = CDIM / 16;   // 16
  const long task = (long)blockIdx.x * 8 + wave;
  const long tpb = (long)ltiles * etiles;
  const int b = (int)(task / tpb);
  if (b >= BATCH) return;
  const int rr = (int)(task % tpb);
  const int e0 = (rr / ltiles) * 16;
  const int l0 = (rr % ltiles) * 64;
  const int col = lane & 15, hh = lane >> 4;

  const float*  Xf = (const float*)Xv + (long)b * CDIM * LDIM;
  const __bf16* Xb = (const __bf16*)Xv + (long)b * CDIM * LDIM;

  v8f acc[4];
  #pragma unroll
  for (int s = 0; s < 4; ++s) {
    v8f z0 = {0, 0, 0, 0, 0, 0, 0, 0};
    acc[s] = z0;
  }

  for (int kc = 0; kc < CDIM; kc += 32) {
    // A fragment: A[m=e_local][k=c]; each half is 8 contiguous bf16.
    v16bf af;
    {
      const __bf16* abase = &Wt[(e0 + col) * CDIM + kc + hh * 8];
      const v8bf alo = ld8(abase);        // k offsets hh*8 + 0..7
      const v8bf ahi = ld8(abase + 16);   // k offsets 16 + hh*8 + 0..7
      #pragma unroll
      for (int i = 0; i < 8; ++i) { af[i] = alo[i]; af[8 + i] = ahi[i]; }
    }
    // B fragments: B[k=c][n=l_local]; k rows stride LDIM (coalesced across lanes).
    v16bf bfm[4];
    #pragma unroll
    for (int s = 0; s < 4; ++s) {
      const int l = l0 + s * 16 + col;
      #pragma unroll
      for (int p = 0; p < 8; ++p) {
        const int k = kc + hh * 8 + 2 * p + ((p >= 4) ? 8 : 0);
        if (INBF16) {
          bfm[s][2 * p]     = Xb[(long)k * LDIM + l];
          bfm[s][2 * p + 1] = Xb[(long)(k + 1) * LDIM + l];
        } else {
          bfm[s][2 * p]     = f2bf(Xf[(long)k * LDIM + l]);
          bfm[s][2 * p + 1] = f2bf(Xf[(long)(k + 1) * LDIM + l]);
        }
      }
    }
    #pragma unroll
    for (int s = 0; s < 4; ++s) { acc[s] = WMMA_BF16(af, bfm[s], acc[s]); }
  }

  const int d = l0 / NPOS;  // 64-wide l tile never straddles a d boundary
  #pragma unroll
  for (int s = 0; s < 4; ++s) {
    const int l = l0 + s * 16 + col;
    #pragma unroll
    for (int r = 0; r < 8; ++r) {
      const int e = e0 + r + 8 * hh;  // D layout: row = r (+8 for hi lanes)
      const float v = acc[s][r] + U[e * 3 + d];
      if (MODE == 0) {
        const int hd = e >> 5, ch = e & 31;
        const int n = l - d * NPOS;
        ((__bf16*)Yv)[(((long)b * HEADS + hd) * NPOS + n) * KDH + ch * 3 + d] = f2bf(v);
      } else if (MODE == 1) {
        ((__bf16*)Yv)[(long)b * CDIM * LDIM + (long)e * LDIM + l] = f2bf(v);
      } else {
        ((float*)Yv)[(long)b * CDIM * LDIM + (long)e * LDIM + l] = v;
      }
    }
  }
}

// ---------------------------------------------------------------------------
// Flash attention, one wave per (b, h, 16-row m-tile).
// Computes S^T[n,m] (softmax dim lives in-lane), online softmax, then
// O^T[cd,m] += Z'[cd,n] * P^T[n,m] with bf16 WMMA. 12 WMMAs / 32-col tile.
// All fragments load as b128 pairs; next tile prefetched via global_prefetch.
// ---------------------------------------------------------------------------
__global__ void __launch_bounds__(256) vn_attn(
    const __bf16* __restrict__ qb, const __bf16* __restrict__ kb,
    const __bf16* __restrict__ zb, __bf16* __restrict__ ob) {
  const int lane = threadIdx.x & 31;
  const int wave = threadIdx.x >> 5;
  const int mtiles = NPOS / 16;  // 128
  const long task = (long)blockIdx.x * 8 + wave;
  const int b = (int)(task / (HEADS * mtiles));
  if (b >= BATCH) return;
  const int rem = (int)(task % (HEADS * mtiles));
  const int h = rem / mtiles;
  const int m0 = (rem % mtiles) * 16;
  const int col = lane & 15, hh = lane >> 4;

  const __bf16* qh = qb + ((long)b * HEADS + h) * NPOS * KDH;  // [m][96]
  const __bf16* kh = kb + ((long)b * HEADS + h) * NPOS * KDH;  // [n][96]
  const __bf16* zh = zb + (long)b * CDIM * LDIM + (long)h * KDH * NPOS;  // [96][2048]

  // Q as B fragments for S^T = K * Q: B[k][m]; two runs of 8 contiguous bf16 per lane (k = hh*8 + 0..7, 16 + hh*8 + 0..7).
  v16bf qf[3];
  #pragma unroll
  for (int kt = 0; kt < 3; ++kt) {
    const __bf16* qbase = &qh[(long)(m0 + col) * KDH + kt * 32 + hh * 8];
    const v8bf qlo = ld8(qbase);
    const v8bf qhi = ld8(qbase + 16);
    #pragma unroll
    for (int i = 0; i < 8; ++i) { qf[kt][i] = qlo[i]; qf[kt][8 + i] = qhi[i]; }
  }

  v8f o[6];
  #pragma unroll
  for (int t = 0; t < 6; ++t) {
    v8f z0 = {0, 0, 0, 0, 0, 0, 0, 0};
    o[t] = z0;
  }
  float M = -3.0e38f, L = 0.0f;
  const float scale = 0.10206207262f;  // 1/sqrt(3*Ch) = 1/sqrt(96)

  for (int n0 = 0; n0 < NPOS; n0 += 32) {
    // Prefetch next tile's K and Z rows while this tile computes.
    {
      const int np = (n0 + 32 < NPOS) ? (n0 + 32) : 0;
      __builtin_prefetch(&kh[(long)(np + col) * KDH], 0, 1);
      __builtin_prefetch(&zh[(long)(col)*NPOS + np], 0, 1);
    }
    v8f s0 = {0, 0, 0, 0, 0, 0, 0, 0};
    v8f s1 = {0, 0, 0, 0, 0, 0, 0, 0};
    #pragma unroll
    for (int kt = 0; kt < 3; ++kt) {
      v16bf a0, a1;  // A = K rows: A[n_local][k], 2 x b128 each
      {
        const __bf16* k0 = &kh[(long)(n0 + col) * KDH + kt * 32 + hh * 8];
        const __bf16* k1 = &kh[(long)(n0 + 16 + col) * KDH + kt * 32 + hh * 8];
        const v8bf a0lo = ld8(k0), a0hi = ld8(k0 + 16);
        const v8bf a1lo = ld8(k1), a1hi = ld8(k1 + 16);
        #pragma unroll
        for (int i = 0; i < 8; ++i) {
          a0[i] = a0lo[i]; a0[8 + i] = a0hi[i];
          a1[i] = a1lo[i]; a1[8 + i] = a1hi[i];
        }
      }
      s0 = WMMA_BF16(a0, qf[kt], s0);
      s1 = WMMA_BF16(a1, qf[kt], s1);
    }

    // Online softmax over n (in-lane regs + one cross-half shuffle per stat).
    float tmax = -3.0e38f;
    #pragma unroll
    for (int r = 0; r < 8; ++r) {
      s0[r] *= scale;
      s1[r] *= scale;
      tmax = fmaxf(tmax, fmaxf(s0[r], s1[r]));
    }
    tmax = fmaxf(tmax, __shfl_xor(tmax, 16, 32));
    const float newM = fmaxf(M, tmax);
    const float alpha = __expf(M - newM);
    float rs = 0.0f;
    #pragma unroll
    for (int r = 0; r < 8; ++r) {
      s0[r] = __expf(s0[r] - newM);
      s1[r] = __expf(s1[r] - newM);
      rs += s0[r] + s1[r];
    }
    rs += __shfl_xor(rs, 16, 32);
    L = L * alpha + rs;
    M = newM;
    #pragma unroll
    for (int t = 0; t < 6; ++t) {
      #pragma unroll
      for (int r = 0; r < 8; ++r) { o[t][r] *= alpha; }  // alpha is per-m = per-lane
    }

    // Transpose P (D layout) -> B fragment of P^T via lane shuffles.
    v16bf pb;
    #pragma unroll
    for (int p = 0; p < 8; ++p) {
      #pragma unroll
      for (int ee = 0; ee < 2; ++ee) {
        const int j = 2 * p + ee;                   // n_local within half-tile
        const int src = col + ((j & 8) ? 16 : 0);   // lane holding row j, col m
        const float v0 = __shfl(s0[j & 7], src, 32);
        const float v1 = __shfl(s1[j & 7], src, 32);
        pb[2 * p + ee] = f2bf(hh ? v1 : v0);
      }
    }

    // O^T[cd,m] += Z'[cd,n] * P^T[n,m], 6 cd-tiles of 16.
    #pragma unroll
    for (int t = 0; t < 6; ++t) {
      v16bf za;  // A[cd_local][k=n_local], 2 x b128
      {
        const __bf16* zbase = &zh[(long)(t * 16 + col) * NPOS + n0 + hh * 16];
        const v8bf zlo = ld8(zbase);
        const v8bf zhi = ld8(zbase + 8);
        #pragma unroll
        for (int i = 0; i < 8; ++i) { za[i] = zlo[i]; za[8 + i] = zhi[i]; }
      }
      o[t] = WMMA_BF16(za, pb, o[t]);
    }
  }

  const float invL = 1.0f / L;  // per-m, per-lane
  #pragma unroll
  for (int t = 0; t < 6; ++t) {
    #pragma unroll
    for (int r = 0; r < 8; ++r) {
      const int cd = t * 16 + r + 8 * hh;
      ob[(long)b * CDIM * LDIM + ((long)h * KDH + cd) * NPOS + m0 + col] =
          f2bf(o[t][r] * invL);
    }
  }
}

// ---------------------------------------------------------------------------
extern "C" void kernel_launch(void* const* d_in, const int* in_sizes, int n_in,
                              void* d_out, int out_size, void* d_ws, size_t ws_size,
                              hipStream_t stream) {
  (void)in_sizes; (void)n_in; (void)out_size; (void)ws_size;
  const float* Q  = (const float*)d_in[0];
  const float* K  = (const float*)d_in[1];
  const float* Z  = (const float*)d_in[2];
  const float* Wq = (const float*)d_in[3]; const float* bq = (const float*)d_in[4];
  const float* Wk = (const float*)d_in[5]; const float* bk = (const float*)d_in[6];
  const float* Wz = (const float*)d_in[7]; const float* bz = (const float*)d_in[8];
  const float* Wo = (const float*)d_in[9]; const float* bo = (const float*)d_in[10];

  char* ws = (char*)d_ws;
  size_t off = 0;
  auto carve = [&](size_t bytes) -> void* {
    void* p = ws + off;
    off = (off + bytes + 255) & ~(size_t)255;
    return p;
  };
  __bf16* Wtq = (__bf16*)carve((size_t)CDIM * CDIM * 2);
  __bf16* Wtk = (__bf16*)carve((size_t)CDIM * CDIM * 2);
  __bf16* Wtz = (__bf16*)carve((size_t)CDIM * CDIM * 2);
  __bf16* Wto = (__bf16*)carve((size_t)CDIM * CDIM * 2);
  float*  Uq  = (float*)carve((size_t)CDIM * 3 * 4);
  float*  Uk  = (float*)carve((size_t)CDIM * 3 * 4);
  float*  Uz  = (float*)carve((size_t)CDIM * 3 * 4);
  float*  Uo  = (float*)carve((size_t)CDIM * 3 * 4);
  __bf16* qbuf = (__bf16*)carve((size_t)BATCH * HEADS * NPOS * KDH * 2);
  __bf16* kbuf = (__bf16*)carve((size_t)BATCH * HEADS * NPOS * KDH * 2);
  __bf16* zbuf = (__bf16*)carve((size_t)BATCH * CDIM * LDIM * 2);
  __bf16* obuf = (__bf16*)carve((size_t)BATCH * CDIM * LDIM * 2);

  vn_prep<<<1, 256, 0, stream>>>(Wq, bq, Wtq, Uq);
  vn_prep<<<1, 256, 0, stream>>>(Wk, bk, Wtk, Uk);
  vn_prep<<<1, 256, 0, stream>>>(Wz, bz, Wtz, Uz);
  vn_prep<<<1, 256, 0, stream>>>(Wo, bo, Wto, Uo);

  const int projBlocks = (BATCH * (CDIM / 16) * (LDIM / 64)) / 8;  // 768
  vn_proj<0, 0><<<projBlocks, 256, 0, stream>>>(Q, Wtq, Uq, qbuf);
  vn_proj<0, 0><<<projBlocks, 256, 0, stream>>>(K, Wtk, Uk, kbuf);
  vn_proj<1, 0><<<projBlocks, 256, 0, stream>>>(Z, Wtz, Uz, zbuf);

  const int attnBlocks = (BATCH * HEADS * (NPOS / 16)) / 8;  // 512
  vn_attn<<<attnBlocks, 256, 0, stream>>>(qbuf, kbuf, zbuf, obuf);

  vn_proj<2, 1><<<projBlocks, 256, 0, stream>>>(obuf, Wto, Uo, d_out);
}
